// FlexQMixer_2551210573965
// MI455X (gfx1250) — matrix operand order repaired and checked by reading, not compiled, not run on hardware
//
#include <hip/hip_runtime.h>
#include <hip/hip_bf16.h>

typedef __attribute__((ext_vector_type(16))) _Float16 v16h;
typedef __attribute__((ext_vector_type(4)))  _Float16 v4h;
typedef __attribute__((ext_vector_type(8)))  float    v8f;
typedef __attribute__((ext_vector_type(4)))  unsigned int v4u;

union FragU { v16h h; v4u u[2]; };

// ---- swizzled-weight workspace layout (halfs) -------------------------------
// Per hyper: fc1 (4 Ksteps x 8 Ntiles), in_w (4 x 24), out_w (4 x 8), fc2 (4 x 2)
// Each fragment = 512 halfs (32x16 B-tile in per-lane WMMA order, 1 KB).
constexpr int FRAG_H   = 512;
constexpr int OFF_FC1  = 0;                 // 32 frags
constexpr int OFF_IN   = 32 * FRAG_H;       // 96 frags
constexpr int OFF_OUT  = 128 * FRAG_H;      // 32 frags
constexpr int OFF_FC2  = 160 * FRAG_H;      // 8 frags
constexpr int PER_H    = 168 * FRAG_H;      // 86016 halfs per hyper
constexpr size_t WS_NEEDED = (size_t)4 * PER_H * sizeof(_Float16);  // 688128 B

// A-fragment (16x32 f16), contiguous k order (B matches): lanes 0-15 hold K {0..15},
// lanes 16-31 hold K {16..31}. Source row-major [row][K], ld in halfs.
__device__ inline v16h load_fragA(const _Float16* p0, int ld, int row0, int k0) {
  const int lane = threadIdx.x & 31;
  const int hl = lane >> 4;
  const _Float16* p = p0 + (row0 + (lane & 15)) * ld + k0 + hl * 16;
  FragU f;
  f.u[0] = *(const v4u*)p;
  f.u[1] = *(const v4u*)(p + 8);
  return f.h;
}

// B-fragment from LDS B^T staging [n][K]: lanes 0-15 K=0..15, lanes 16-31 K=16..31.
__device__ inline v16h load_fragB(const _Float16* p0, int ld, int row0, int k0) {
  const int lane = threadIdx.x & 31;
  const int hl = lane >> 4;
  const _Float16* p = p0 + (row0 + (lane & 15)) * ld + k0 + hl * 16;
  FragU f;
  f.u[0] = *(const v4u*)p;
  f.u[1] = *(const v4u*)(p + 8);
  return f.h;
}

// B-fragment from pre-swizzled global memory: lane reads 32 contiguous bytes.
__device__ inline v16h load_fragB_g(const _Float16* frag) {
  const _Float16* p = frag + (threadIdx.x & 31) * 16;
  FragU f;
  f.u[0] = *(const v4u*)p;
  f.u[1] = *(const v4u*)(p + 8);
  return f.h;
}

__device__ inline v8f wmma_f16(v16h a, v16h b, v8f c) {
  return __builtin_amdgcn_wmma_f32_16x16x32_f16(false, a, false, b, (short)0, c,
                                                false, false);
}

// Fallback-path staging: f32 weight K-slab -> LDS f16 [ncols][32] transposed.
__device__ inline void stage_wT(const float* __restrict__ g, int ldn, int nbase,
                                int ncols, int k0, _Float16* sW) {
  const int tid = threadIdx.x;
  const int total = 32 * ncols;
  for (int idx = tid; idx < total; idx += 256) {
    int kk = idx / ncols;
    int n = idx - kk * ncols;
    sW[n * 32 + kk] = (_Float16)g[(size_t)(k0 + kk) * ldn + nbase + n];
  }
}

// ---- one-time prepass: convert f32 weights to f16 in WMMA B-fragment order --
__global__ __launch_bounds__(256) void swizzle_weights_kernel(
    const float* __restrict__ fc1_w, const float* __restrict__ in_w,
    const float* __restrict__ out_w, const float* __restrict__ fc2_w,
    _Float16* __restrict__ wsw) {
  const int frag = blockIdx.x * 8 + (threadIdx.x >> 5);  // 672 total
  const int lane = threadIdx.x & 31;
  const int h = frag / 168;
  int f = frag - h * 168;
  const float* g;
  int ldn, ntiles;
  size_t off;
  if (f < 32)       { g = fc1_w + (size_t)h * 128 * 128; ldn = 128; ntiles = 8;  off = OFF_FC1; }
  else if (f < 128) { g = in_w  + (size_t)h * 128 * 384; ldn = 384; ntiles = 24; off = OFF_IN;  f -= 32; }
  else if (f < 160) { g = out_w + (size_t)h * 128 * 128; ldn = 128; ntiles = 8;  off = OFF_OUT; f -= 128; }
  else              { g = fc2_w + (size_t)h * 128 * 32;  ldn = 32;  ntiles = 2;  off = OFF_FC2; f -= 160; }
  const int ks = f / ntiles;
  const int nt = f - ks * ntiles;
  const int n = nt * 16 + (lane & 15);
  const int kbase = ks * 32 + (lane >> 4) * 16;
  union { _Float16 e[16]; v4u u[2]; } t;
#pragma unroll
  for (int i = 0; i < 16; ++i)
    t.e[i] = (_Float16)g[(size_t)(kbase + i) * ldn + n];
  _Float16* dst = wsw + (size_t)h * PER_H + off + (size_t)f * FRAG_H + lane * 16;
  *(v4u*)dst = t.u[0];
  *(v4u*)(dst + 8) = t.u[1];
}

// ---- main kernel ------------------------------------------------------------
template <bool PRE>
__global__ __launch_bounds__(256) void flexqmix_kernel(
    const float* __restrict__ entities, const float* __restrict__ agent_qs,
    const int* __restrict__ entity_mask, const float* __restrict__ fc1_w,
    const float* __restrict__ fc1_b, const float* __restrict__ in_w,
    const float* __restrict__ out_w, const float* __restrict__ out_b,
    const float* __restrict__ fc2_w, const float* __restrict__ fc2_b,
    const _Float16* __restrict__ wsw, float* __restrict__ out) {
  constexpr int NE = 64, D = 128, H = 128, E = 32, NA = 16;

  __shared__ __align__(16) _Float16 sE[NE * D];
  __shared__ __align__(16) _Float16 sX1[NE * H];
  __shared__ __align__(16) _Float16 sQ[NA * H];
  __shared__ __align__(16) _Float16 sK[NE * H];
  __shared__ __align__(16) _Float16 sVT[H * NE];
  __shared__ __align__(16) float    sLg[4 * NA * NE];
  __shared__ __align__(16) _Float16 sWgt[4 * NA * NE];
  __shared__ __align__(16) _Float16 sAttn[NA * H];
  __shared__ __align__(16) _Float16 sY[NA * H];
  __shared__ __align__(16) _Float16 sW[H * 32];  // fallback staging only
  __shared__ __align__(16) float    sX3[NA * E];
  __shared__ __align__(16) float    sW1[NA * E];
  __shared__ float sB1[E], sWf[E], sRed[E], sV;
  __shared__ float sQs[NA];
  __shared__ int   sM[NE];

  const int tid = threadIdx.x;
  const int wave = tid >> 5;
  const int lane = tid & 31;
  const int b = blockIdx.x;

  // ---- load entities (vectorized), masks, qs ----
  {
    const float4* eb = (const float4*)(entities + (size_t)b * NE * D);
#pragma unroll
    for (int i = 0; i < 8; ++i) {
      float4 v = eb[tid + i * 256];
      v4h hv = {(_Float16)v.x, (_Float16)v.y, (_Float16)v.z, (_Float16)v.w};
      *(v4h*)&sE[(tid + i * 256) * 4] = hv;
    }
    if (tid < NE) sM[tid] = entity_mask[(size_t)b * NE + tid];
    if (tid < NA) sQs[tid] = agent_qs[(size_t)b * NA + tid];
  }
  __syncthreads();

  for (int h = 0; h < 4; ++h) {
    const float* w1g = fc1_w + (size_t)h * D * H;
    const float* b1g = fc1_b + (size_t)h * H;
    const float* wig = in_w + (size_t)h * H * 384;
    const float* wog = out_w + (size_t)h * H * H;
    const float* bog = out_b + (size_t)h * H;
    const float* w2g = fc2_w + (size_t)h * H * E;
    const float* b2g = fc2_b + (size_t)h * E;

    if constexpr (PRE) {
      // pull next hyper's swizzled weights toward the WGP while we compute
      if (h < 3) {
        const _Float16* nxt = wsw + (size_t)(h + 1) * PER_H;
        for (int i = tid; i < PER_H / 64; i += 256)
          __builtin_prefetch(nxt + (size_t)i * 64, 0, 3);
      }
    }

    // ---------------- x1 = relu(E @ W1 + b1)  [64x128] ----------------
    {
      const int mt = wave & 3;
      const int ng = (wave >> 2) * 4;
      v8f acc[4] = {};
      if constexpr (PRE) {
        const _Float16* wf = wsw + (size_t)h * PER_H + OFF_FC1;
#pragma unroll
        for (int ks = 0; ks < 4; ++ks) {
          v16h a = load_fragA(sE, D, mt * 16, ks * 32);
#pragma unroll
          for (int j = 0; j < 4; ++j)
            acc[j] = wmma_f16(
                a, load_fragB_g(wf + (size_t)(ks * 8 + ng + j) * FRAG_H), acc[j]);
        }
      } else {
        for (int ks = 0; ks < 4; ++ks) {
          __syncthreads();
          stage_wT(w1g, H, 0, H, ks * 32, sW);
          __syncthreads();
          v16h a = load_fragA(sE, D, mt * 16, ks * 32);
#pragma unroll
          for (int j = 0; j < 4; ++j)
            acc[j] = wmma_f16(a, load_fragB(sW, 32, (ng + j) * 16, 0), acc[j]);
        }
      }
#pragma unroll
      for (int j = 0; j < 4; ++j) {
        int n = (ng + j) * 16 + (lane & 15);
        float bias = b1g[n];
#pragma unroll
        for (int r = 0; r < 8; ++r) {
          int m = mt * 16 + r + 8 * (lane >> 4);
          float v = acc[j][r] + bias;
          sX1[m * H + n] = (_Float16)(v > 0.f ? v : 0.f);
        }
      }
    }
    __syncthreads();

    // ---------------- qkv = x1 @ Win  [64x384], 3 groups of 128 cols ----
    for (int grp = 0; grp < 3; ++grp) {
      const int mt = wave & 3;
      const int ng = (wave >> 2) * 4;
      v8f acc[4] = {};
      if constexpr (PRE) {
        const _Float16* wf = wsw + (size_t)h * PER_H + OFF_IN;
#pragma unroll
        for (int ks = 0; ks < 4; ++ks) {
          v16h a = load_fragA(sX1, H, mt * 16, ks * 32);
#pragma unroll
          for (int j = 0; j < 4; ++j)
            acc[j] = wmma_f16(
                a,
                load_fragB_g(wf + (size_t)(ks * 24 + grp * 8 + ng + j) * FRAG_H),
                acc[j]);
        }
      } else {
        for (int ks = 0; ks < 4; ++ks) {
          __syncthreads();
          stage_wT(wig, 384, grp * 128, 128, ks * 32, sW);
          __syncthreads();
          v16h a = load_fragA(sX1, H, mt * 16, ks * 32);
#pragma unroll
          for (int j = 0; j < 4; ++j)
            acc[j] = wmma_f16(a, load_fragB(sW, 32, (ng + j) * 16, 0), acc[j]);
        }
      }
#pragma unroll
      for (int j = 0; j < 4; ++j) {
        int c = grp * 128 + (ng + j) * 16 + (lane & 15);
#pragma unroll
        for (int r = 0; r < 8; ++r) {
          int m = mt * 16 + r + 8 * (lane >> 4);
          _Float16 hv = (_Float16)acc[j][r];
          if (c < 128) {
            if (m < NA) sQ[m * H + c] = hv;
          } else if (c < 256) {
            sK[m * H + (c - 128)] = hv;
          } else {
            sVT[(c - 256) * NE + m] = hv;  // v stored transposed
          }
        }
      }
    }
    __syncthreads();

    // ---------------- logits = q k^T / sqrt(32)  (wave per head) --------
    if (wave < 4) {
      const int head = wave;
      v16h aq = load_fragA(sQ, H, 0, head * 32);
#pragma unroll
      for (int nt = 0; nt < 4; ++nt) {
        v16h bk = load_fragB(sK, H, nt * 16, head * 32);
        v8f c = {};
        c = wmma_f16(aq, bk, c);
        int n = nt * 16 + (lane & 15);
#pragma unroll
        for (int r = 0; r < 8; ++r) {
          int m = r + 8 * (lane >> 4);
          sLg[head * NA * NE + m * NE + n] = c[r] * 0.17677669529663687f;
        }
      }
    }
    __syncthreads();

    // ---------------- masked softmax (64 rows, one thread each) ---------
    if (tid < 64) {
      int head = tid >> 4, m = tid & 15;
      const float* row = sLg + head * NA * NE + m * NE;
      bool am = sM[m] != 0;
      float mx = -3.0e38f;
      for (int j = 0; j < NE; ++j)
        if (!am && sM[j] == 0) mx = fmaxf(mx, row[j]);
      float sum = 0.f;
      for (int j = 0; j < NE; ++j)
        if (!am && sM[j] == 0) sum += __expf(row[j] - mx);
      float inv = sum > 0.f ? 1.f / sum : 0.f;  // fully-masked row -> zeros
      for (int j = 0; j < NE; ++j) {
        float w = (!am && sM[j] == 0) ? __expf(row[j] - mx) * inv : 0.f;
        sWgt[head * NA * NE + m * NE + j] = (_Float16)w;
      }
    }
    __syncthreads();

    // ---------------- attn = w @ v  (wave per head) ----------------------
    if (wave < 4) {
      const int head = wave;
#pragma unroll
      for (int nt = 0; nt < 2; ++nt) {
        v8f c = {};
#pragma unroll
        for (int ks = 0; ks < 2; ++ks) {
          v16h aw = load_fragA(sWgt + head * NA * NE, NE, 0, ks * 32);
          v16h bv = load_fragB(sVT, NE, head * 32 + nt * 16, ks * 32);
          c = wmma_f16(aw, bv, c);
        }
        int d = head * 32 + nt * 16 + (lane & 15);
#pragma unroll
        for (int r = 0; r < 8; ++r) {
          int m = r + 8 * (lane >> 4);
          sAttn[m * H + d] = (_Float16)c[r];
        }
      }
    }
    __syncthreads();

    // ---------------- out = attn @ Wout + b, post-mask  [16x128] ---------
    {
      v8f acc = {};
      if constexpr (PRE) {
        const _Float16* wf = wsw + (size_t)h * PER_H + OFF_OUT;
#pragma unroll
        for (int ks = 0; ks < 4; ++ks) {
          v16h a = load_fragA(sAttn, H, 0, ks * 32);
          acc = wmma_f16(a, load_fragB_g(wf + (size_t)(ks * 8 + wave) * FRAG_H),
                         acc);
        }
      } else {
        for (int ks = 0; ks < 4; ++ks) {
          __syncthreads();
          stage_wT(wog, H, 0, H, ks * 32, sW);
          __syncthreads();
          v16h a = load_fragA(sAttn, H, 0, ks * 32);
          acc = wmma_f16(a, load_fragB(sW, 32, wave * 16, 0), acc);
        }
      }
      int n = wave * 16 + (lane & 15);
      float bias = bog[n];
#pragma unroll
      for (int r = 0; r < 8; ++r) {
        int m = r + 8 * (lane >> 4);
        float v = acc[r] + bias;
        if (sM[m] != 0) v = 0.f;
        sY[m * H + n] = (_Float16)v;
      }
    }
    __syncthreads();

    // ---------------- x3 = y @ W2 + b2, post-mask  [16x32] ---------------
    {
      v8f acc = {};
      if constexpr (PRE) {
        if (wave < 2) {
          const _Float16* wf = wsw + (size_t)h * PER_H + OFF_FC2;
#pragma unroll
          for (int ks = 0; ks < 4; ++ks) {
            v16h a = load_fragA(sY, H, 0, ks * 32);
            acc = wmma_f16(
                a, load_fragB_g(wf + (size_t)(ks * 2 + wave) * FRAG_H), acc);
          }
        }
      } else {
        for (int ks = 0; ks < 4; ++ks) {
          __syncthreads();
          stage_wT(w2g, E, 0, E, ks * 32, sW);
          __syncthreads();
          if (wave < 2) {
            v16h a = load_fragA(sY, H, 0, ks * 32);
            acc = wmma_f16(a, load_fragB(sW, 32, wave * 16, 0), acc);
          }
        }
      }
      if (wave < 2) {
        int n = wave * 16 + (lane & 15);
        float bias = b2g[n];
#pragma unroll
        for (int r = 0; r < 8; ++r) {
          int m = r + 8 * (lane >> 4);
          float v = acc[r] + bias;
          if (sM[m] != 0) v = 0.f;
          sX3[m * E + n] = v;
        }
      }
    }
    __syncthreads();

    // ---------------- per-hyper reductions -------------------------------
    if (tid < 32) {
      int e = tid;
      if (h == 0) {
        for (int a = 0; a < NA; ++a) sW1[a * E + e] = fabsf(sX3[a * E + e]);
      } else {
        float s = 0.f;
        for (int a = 0; a < NA; ++a) s += sX3[a * E + e];
        s *= (1.f / 16.f);
        if (h == 1) sB1[e] = fabsf(s);
        else if (h == 2) sWf[e] = fabsf(s);
        else sRed[e] = s;  // raw column mean; abs after full mean
      }
    }
    __syncthreads();
    if (h == 3 && tid == 0) {
      float t = 0.f;
      for (int e = 0; e < E; ++e) t += sRed[e];
      sV = fabsf(t * (1.f / 32.f));
    }
    __syncthreads();
    (void)w1g; (void)wig; (void)wog; (void)w2g;
  }  // hyper loop

  // ---- final mix: elu(qs @ |w1| + b1) . wf + v ----
  if (tid < 32) {
    int e = tid;
    float acc = sB1[e];
    for (int a = 0; a < NA; ++a) acc += sQs[a] * sW1[a * E + e];
    float hid = acc > 0.f ? acc : (__expf(acc) - 1.f);  // elu(alpha=1)
    sRed[e] = hid * sWf[e];
  }
  __syncthreads();
  if (tid == 0) {
    float y = sV;
    for (int e = 0; e < E; ++e) y += sRed[e];
    out[b] = y;
  }
}

extern "C" void kernel_launch(void* const* d_in, const int* in_sizes, int n_in,
                              void* d_out, int out_size, void* d_ws,
                              size_t ws_size, hipStream_t stream) {
  const float* entities = (const float*)d_in[0];
  const float* agent_qs = (const float*)d_in[1];
  const int* entity_mask = (const int*)d_in[2];
  const float* fc1_w = (const float*)d_in[3];
  const float* fc1_b = (const float*)d_in[4];
  const float* in_w = (const float*)d_in[5];
  const float* out_w = (const float*)d_in[6];
  const float* out_b = (const float*)d_in[7];
  const float* fc2_w = (const float*)d_in[8];
  const float* fc2_b = (const float*)d_in[9];
  float* out = (float*)d_out;

  const int B = in_sizes[1] / 16;  // BS*T = 8192

  if (ws_size >= WS_NEEDED) {
    _Float16* wsw = (_Float16*)d_ws;
    swizzle_weights_kernel<<<dim3(84), dim3(256), 0, stream>>>(
        fc1_w, in_w, out_w, fc2_w, wsw);
    flexqmix_kernel<true><<<dim3(B), dim3(256), 0, stream>>>(
        entities, agent_qs, entity_mask, fc1_w, fc1_b, in_w, out_w, out_b,
        fc2_w, fc2_b, wsw, out);
  } else {
    flexqmix_kernel<false><<<dim3(B), dim3(256), 0, stream>>>(
        entities, agent_qs, entity_mask, fc1_w, fc1_b, in_w, out_w, out_b,
        fc2_w, fc2_b, nullptr, out);
  }
}
